// NanoChatAttention_10806137717006
// MI455X (gfx1250) — compile-verified, matrix operand order checked by reading, not run on hardware
//
#include <hip/hip_runtime.h>
#include <hip/hip_bf16.h>
#include <math.h>
#include <stdint.h>

// ---------------------------------------------------------------------------
// NanoChat attention block for MI455X (gfx1250, wave32).
// bf16 operands everywhere (packed once), fp32 accumulation, WMMA
// 16x16x32_bf16 for all matmuls, TDM (tensor_load_to_lds) staging of KV
// tiles in the flash-attention kernel.
// ---------------------------------------------------------------------------

#define DEV __device__ __forceinline__

typedef __attribute__((ext_vector_type(16))) __bf16 v16bf;
typedef __attribute__((ext_vector_type(8)))  __bf16 v8bf;
typedef __attribute__((ext_vector_type(8)))  float  v8f;
typedef __attribute__((ext_vector_type(4)))  unsigned int u32x4;
typedef __attribute__((ext_vector_type(8)))  int i32x8;
typedef __attribute__((ext_vector_type(4)))  int i32x4;

constexpr int Bc   = 2;
constexpr int Tc   = 2048;
constexpr int Dc   = 2048;
constexpr int NHc  = 16;
constexpr int NKVc = 4;
constexpr int HDc  = 128;
constexpr int NREP = NHc / NKVc;

DEV v8f zero8() {
  v8f z;
#pragma unroll
  for (int i = 0; i < 8; ++i) z[i] = 0.0f;
  return z;
}

// Copy 8 bf16 (16B vector load: global_load_b128 / ds_load_b128) into
// halves of a v16bf WMMA operand.  A/B 16-bit operands are two contiguous
// runs of 8 K-values per lane: k = 8*half + i and k = 16 + 8*half + i.
DEV void cp8lo(v16bf& a, const __bf16* p) {
  v8bf t = *(const v8bf*)p;
#pragma unroll
  for (int i = 0; i < 8; ++i) a[i] = t[i];
}
DEV void cp8hi(v16bf& a, const __bf16* p) {
  v8bf t = *(const v8bf*)p;
#pragma unroll
  for (int i = 0; i < 8; ++i) a[8 + i] = t[i];
}

// ---------------------------------------------------------------------------
// TDM: 1-D contiguous DMA of n2B bf16 elements from global to LDS.
// D# built per CDNA5 ISA 8.3-8.5:  group0 = {count=1 | lds_addr |
// global_addr[31:0] | global_addr[56:32] + type=2},  group1 = {data_size=2B,
// tensor_dim0 = tile_dim0 = n2B, tensor_dim1 = 1, stride0 = n2B}.
// 6-arg clang-23 builtin: (u32x4 g0, i32x8 g1, i32x4, i32x4, i32x8, i32 cpol)
// ---------------------------------------------------------------------------
DEV void tdm_load_1d(unsigned lds_off, const void* gsrc, unsigned n2B) {
  const uint64_t ga = (uint64_t)(uintptr_t)gsrc;
  u32x4 g0;
  g0[0] = 1u;                                            // count=1 (valid user D#)
  g0[1] = lds_off;                                       // lds_addr (bytes)
  g0[2] = (unsigned)ga;                                  // global_addr[31:0]
  g0[3] = (unsigned)((ga >> 32) & 0x01FFFFFFu) | 0x80000000u;  // [56:32] | type=2
  i32x8 g1;
  g1[0] = (int)(1u << 16);                               // data_size=1 -> 2 bytes
  g1[1] = (int)((n2B & 0xFFFFu) << 16);                  // tensor_dim0[15:0]
  g1[2] = (int)(((n2B >> 16) & 0xFFFFu) | (1u << 16));   // td0[31:16] | tensor_dim1=1
  g1[3] = (int)((n2B & 0xFFFFu) << 16);                  // tile_dim0 = n2B
  g1[4] = 0;                                             // tile_dim1/2 unused
  g1[5] = (int)n2B;                                      // tensor_dim0_stride lo
  g1[6] = 0;
  g1[7] = 0;
  i32x4 z4;
  z4[0] = 0; z4[1] = 0; z4[2] = 0; z4[3] = 0;
  i32x8 z8;
#pragma unroll
  for (int i = 0; i < 8; ++i) z8[i] = 0;
  __builtin_amdgcn_tensor_load_to_lds(g0, g1, z4, z4, z8, 0);
}

// ---------------------------------------------------------------------------
// Kernel 0: fp32 -> bf16 pack (hidden states & weights), 4 elems/thread.
// ---------------------------------------------------------------------------
__global__ void pack_bf16(const float* __restrict__ x, __bf16* __restrict__ y,
                          int n) {
  const int i = (blockIdx.x * 256 + threadIdx.x) * 4;
  if (i < n) {
    const float4 v = *(const float4*)(x + i);
    y[i + 0] = (__bf16)v.x;
    y[i + 1] = (__bf16)v.y;
    y[i + 2] = (__bf16)v.z;
    y[i + 3] = (__bf16)v.w;
  }
}

// ---------------------------------------------------------------------------
// Kernel 1: Y[b,h,t,hd] = sum_d X[m,d]*W[n,d], bf16 in, fp32 or bf16 out.
// 8 waves/block; wave computes 16(M) x 64(N); K swept in 32s.
// grid = (M/16, Ncols/512)
// ---------------------------------------------------------------------------
template <bool BF16OUT>
__global__ __launch_bounds__(256) void gemm_proj(const __bf16* __restrict__ X,
                                                 const __bf16* __restrict__ W,
                                                 float* __restrict__ Yf,
                                                 __bf16* __restrict__ Yb, int H) {
  const int w    = threadIdx.x >> 5;
  const int lane = threadIdx.x & 31;
  const int half = lane >> 4;
  const int l16  = lane & 15;
  const int M0   = blockIdx.x * 16;
  const int N0   = blockIdx.y * 512 + w * 64;

  v8f acc[4];
#pragma unroll
  for (int i = 0; i < 4; ++i) acc[i] = zero8();

  const __bf16* Xrow = X + (size_t)(M0 + l16) * Dc;

  for (int kk = 0; kk < Dc; kk += 32) {
    v16bf a;
    cp8lo(a, Xrow + kk + 8 * half);
    cp8hi(a, Xrow + kk + 16 + 8 * half);
    v16bf bm[4];
#pragma unroll
    for (int nt = 0; nt < 4; ++nt) {
      const __bf16* Wrow = W + (size_t)(N0 + nt * 16 + l16) * Dc + kk;
      cp8lo(bm[nt], Wrow + 8 * half);
      cp8hi(bm[nt], Wrow + 16 + 8 * half);
    }
#pragma unroll
    for (int nt = 0; nt < 4; ++nt)
      acc[nt] = __builtin_amdgcn_wmma_f32_16x16x32_bf16(
          false, a, false, bm[nt], (short)0, acc[nt], false, false);
  }

#pragma unroll
  for (int nt = 0; nt < 4; ++nt) {
    const int n  = N0 + nt * 16 + l16;
    const int h  = n / HDc;
    const int hd = n % HDc;
#pragma unroll
    for (int r = 0; r < 8; ++r) {
      const int m = M0 + r + 8 * half;        // C layout: row = r + 8*half
      const int b = m / Tc, t = m % Tc;
      const size_t o = (((size_t)b * H + h) * Tc + t) * HDc + hd;
      if (BF16OUT) Yb[o] = (__bf16)acc[nt][r];
      else         Yf[o] = acc[nt][r];
    }
  }
}

// ---------------------------------------------------------------------------
// Kernel 2: RoPE + RMSNorm, fp32 in -> bf16 out (outscale folds 1/sqrt(hd)
// for Q).  One 128-thread block per (b*H+h, t).
// ---------------------------------------------------------------------------
__global__ void rope_rms_bf(const float* __restrict__ X,
                            __bf16* __restrict__ Y,
                            const float* __restrict__ cosT,
                            const float* __restrict__ sinT,
                            float outscale) {
  const int t  = blockIdx.x % Tc;
  const int bh = blockIdx.x / Tc;
  const float* row = X + ((size_t)bh * Tc + t) * HDc;
  __bf16* orow = Y + ((size_t)bh * Tc + t) * HDc;
  const int hd = threadIdx.x;     // 128 threads

  __shared__ float xs[HDc];
  __shared__ float red[4];

  xs[hd] = row[hd];
  __syncthreads();

  const float x   = xs[hd];
  const float rot = (hd < 64) ? -xs[hd + 64] : xs[hd - 64];
  float val = x * cosT[t * HDc + hd] + rot * sinT[t * HDc + hd];

  float ss = val * val;
#pragma unroll
  for (int msk = 16; msk >= 1; msk >>= 1) ss += __shfl_xor(ss, msk, 32);
  if ((hd & 31) == 0) red[hd >> 5] = ss;
  __syncthreads();
  const float tot = red[0] + red[1] + red[2] + red[3];
  val *= rsqrtf(tot * (1.0f / 128.0f) + 1.1920929e-07f);  // eps = FLT_EPSILON
  orow[hd] = (__bf16)(val * outscale);
}

// ---------------------------------------------------------------------------
// Kernel 3: causal flash attention, bf16 Q/K/V -> bf16 O.
// 256 threads (8 waves) per (b, h, 128-query tile).  KV tiles (32 keys) are
// staged into LDS by the Tensor Data Mover (1-D D#, 8KB contiguous), V is
// transposed LDS->LDS for the PV B-operand.  Online softmax fp32.
// __launch_bounds__(256, 1): let the allocator use the full VGPR file so the
// 64-VGPR O accumulator + 32-VGPR Q tile never spill to scratch.
// ---------------------------------------------------------------------------
__global__ __launch_bounds__(256, 1) void attention(const __bf16* __restrict__ Qg,
                                                    const __bf16* __restrict__ Kg,
                                                    const __bf16* __restrict__ Vg,
                                                    __bf16* __restrict__ Og) {
  __shared__ __bf16 Ksh[32 * 128];     // [key][d]     (TDM destination)
  __shared__ __bf16 Vrm[32 * 128];     // [key][d]     (TDM destination)
  __shared__ __bf16 Vt[128 * 32];      // [d][key]     (transposed for PV)
  __shared__ __bf16 Psh[8][16 * 32];   // per-wave P scratch [m][key]

  const int tid  = threadIdx.x;
  const int w    = tid >> 5;
  const int lane = tid & 31;
  const int half = lane >> 4;
  const int l16  = lane & 15;

  const int qt  = blockIdx.x % (Tc / 128);
  const int bh  = blockIdx.x / (Tc / 128);
  const int h   = bh % NHc;
  const int b   = bh / NHc;
  const int kvh = h / NREP;
  const int qbase = qt * 128;
  const int rowbase = qbase + w * 16 + 8 * half;   // row of oacc[*][r=0]

  // Wave's Q tile (16 x 128) as four bf16 A operands (1/sqrt(hd) pre-folded).
  const __bf16* Qrow =
      Qg + (((size_t)b * NHc + h) * Tc + (qbase + w * 16 + l16)) * HDc;
  v16bf aQ[4];
#pragma unroll
  for (int c = 0; c < 4; ++c) {
    cp8lo(aQ[c], Qrow + 32 * c + 8 * half);
    cp8hi(aQ[c], Qrow + 32 * c + 16 + 8 * half);
  }

  v8f   oacc[8];
  float mrun[8], lrun[8];
#pragma unroll
  for (int i = 0; i < 8; ++i) { oacc[i] = zero8(); mrun[i] = -1e30f; lrun[i] = 0.0f; }

  const __bf16* Kbase = Kg + ((size_t)b * NKVc + kvh) * Tc * HDc;
  const __bf16* Vbase = Vg + ((size_t)b * NKVc + kvh) * Tc * HDc;
  const unsigned ldsK = (unsigned)(uintptr_t)&Ksh[0];
  const unsigned ldsV = (unsigned)(uintptr_t)&Vrm[0];

  const int nkt = qbase / 32 + 4;        // KV tiles needed by this block
  for (int kt = 0; kt < nkt; ++kt) {
    __syncthreads();                      // previous tile fully consumed
    if (w == 0) {
      // 8KB contiguous K and V tiles via Tensor Data Mover.
      tdm_load_1d(ldsK, Kbase + (size_t)kt * 32 * HDc, 32 * HDc);
      tdm_load_1d(ldsV, Vbase + (size_t)kt * 32 * HDc, 32 * HDc);
      __builtin_amdgcn_s_wait_tensorcnt(0);
    }
    __syncthreads();                      // K/Vrm tiles resident in LDS

    // Transpose V in LDS: Vrm[key][d] -> Vt[d][key] (8 bf16 per thread x2).
#pragma unroll
    for (int rnd = 0; rnd < 2; ++rnd) {
      const int idx = (rnd * 256 + tid) * 8;
      const int key = idx >> 7;
      const int d   = idx & 127;
      v8bf vv = *(const v8bf*)&Vrm[key * 128 + d];
#pragma unroll
      for (int j = 0; j < 8; ++j) Vt[(d + j) * 32 + key] = vv[j];
    }
    __syncthreads();                      // Vt ready

    // Wave-uniform causal skip: all of this wave's rows precede the tile.
    if (kt * 32 > qbase + w * 16 + 15) continue;

    // ---- scores: S = Q . K^T, two 16x16 C tiles (32 keys) ----
    v8f s[2];
    s[0] = zero8(); s[1] = zero8();
#pragma unroll
    for (int nt = 0; nt < 2; ++nt) {
      const __bf16* Krow = &Ksh[(nt * 16 + l16) * 128];
#pragma unroll
      for (int c = 0; c < 4; ++c) {
        v16bf bm;
        cp8lo(bm, Krow + 32 * c + 8 * half);
        cp8hi(bm, Krow + 32 * c + 16 + 8 * half);
        s[nt] = __builtin_amdgcn_wmma_f32_16x16x32_bf16(
            false, aQ[c], false, bm, (short)0, s[nt], false, false);
      }
    }

    // ---- causal mask: only diagonal-crossing tiles need it (wave-uniform) --
    if (kt * 32 + 31 > qbase + w * 16) {
#pragma unroll
      for (int nt = 0; nt < 2; ++nt) {
        const int n = kt * 32 + nt * 16 + l16;
#pragma unroll
        for (int r = 0; r < 8; ++r) {
          if (n > rowbase + r) s[nt][r] = -1e30f;
        }
      }
    }

    // ---- online softmax (row's 16 columns live in one half-wave) ----
#pragma unroll
    for (int r = 0; r < 8; ++r) {
      float tm = fmaxf(s[0][r], s[1][r]);
#pragma unroll
      for (int msk = 1; msk < 16; msk <<= 1) tm = fmaxf(tm, __shfl_xor(tm, msk, 32));
      const float nm    = fmaxf(mrun[r], tm);
      const float alpha = __expf(mrun[r] - nm);
      mrun[r] = nm;
      const float p0 = __expf(s[0][r] - nm);
      const float p1 = __expf(s[1][r] - nm);
      s[0][r] = p0; s[1][r] = p1;
      float rs = p0 + p1;
#pragma unroll
      for (int msk = 1; msk < 16; msk <<= 1) rs += __shfl_xor(rs, msk, 32);
      lrun[r] = lrun[r] * alpha + rs;
#pragma unroll
      for (int dt = 0; dt < 8; ++dt) oacc[dt][r] *= alpha;
    }

    // ---- P: C-layout fp32 -> A-layout bf16 via per-wave LDS scratch ----
    __bf16* Pw = &Psh[w][0];
#pragma unroll
    for (int nt = 0; nt < 2; ++nt)
#pragma unroll
      for (int r = 0; r < 8; ++r)
        Pw[(r + 8 * half) * 32 + nt * 16 + l16] = (__bf16)s[nt][r];
    __builtin_amdgcn_wave_barrier();
    asm volatile("s_wait_dscnt 0" ::: "memory");   // same-wave LDS RAW

    v16bf aP;
    {
      const __bf16* prow = Pw + l16 * 32;
      cp8lo(aP, prow + 8 * half);
      cp8hi(aP, prow + 16 + 8 * half);
    }

    // ---- O += P . V  (K-dim = 32 keys, 8 N-tiles over hd) ----
#pragma unroll
    for (int dt = 0; dt < 8; ++dt) {
      const __bf16* vcol = &Vt[(dt * 16 + l16) * 32];
      v16bf bm;
      cp8lo(bm, vcol + 8 * half);
      cp8hi(bm, vcol + 16 + 8 * half);
      oacc[dt] = __builtin_amdgcn_wmma_f32_16x16x32_bf16(
          false, aP, false, bm, (short)0, oacc[dt], false, false);
    }
  }

  // ---- epilogue: O / rowsum -> bf16 Og[b,h,t,hd] ----
  __bf16* Ob = Og + ((size_t)b * NHc + h) * Tc * HDc;
#pragma unroll
  for (int r = 0; r < 8; ++r) {
    const float inv = 1.0f / lrun[r];
    const int t = rowbase + r;
#pragma unroll
    for (int dt = 0; dt < 8; ++dt)
      Ob[(size_t)t * HDc + dt * 16 + l16] = (__bf16)(oacc[dt][r] * inv);
  }
}

// ---------------------------------------------------------------------------
// Kernel 4: out[m,n] = sum_d O[b, d/128, t, d%128] * Wo[n,d], bf16 in,
// fp32 out row-major.  8-element runs never cross a head boundary (8 | 128).
// grid = (M/16, 2048/512)
// ---------------------------------------------------------------------------
__global__ __launch_bounds__(256) void gemm_out(const __bf16* __restrict__ Og,
                                                const __bf16* __restrict__ W,
                                                float* __restrict__ out) {
  const int w    = threadIdx.x >> 5;
  const int lane = threadIdx.x & 31;
  const int half = lane >> 4;
  const int l16  = lane & 15;
  const int M0   = blockIdx.x * 16;
  const int N0   = blockIdx.y * 512 + w * 64;

  v8f acc[4];
#pragma unroll
  for (int i = 0; i < 4; ++i) acc[i] = zero8();

  const int m = M0 + l16;
  const int b = m / Tc, t = m % Tc;
  const __bf16* Ob = Og + (size_t)b * NHc * Tc * HDc + (size_t)t * HDc;

  for (int kk = 0; kk < Dc; kk += 32) {
    v16bf a;
    {
      const int d0 = kk + 8 * half;
      cp8lo(a, Ob + (size_t)(d0 >> 7) * Tc * HDc + (d0 & 127));
      const int d1 = kk + 16 + 8 * half;
      cp8hi(a, Ob + (size_t)(d1 >> 7) * Tc * HDc + (d1 & 127));
    }
    v16bf bm[4];
#pragma unroll
    for (int nt = 0; nt < 4; ++nt) {
      const __bf16* Wrow = W + (size_t)(N0 + nt * 16 + l16) * Dc + kk;
      cp8lo(bm[nt], Wrow + 8 * half);
      cp8hi(bm[nt], Wrow + 16 + 8 * half);
    }
#pragma unroll
    for (int nt = 0; nt < 4; ++nt)
      acc[nt] = __builtin_amdgcn_wmma_f32_16x16x32_bf16(
          false, a, false, bm[nt], (short)0, acc[nt], false, false);
  }

#pragma unroll
  for (int nt = 0; nt < 4; ++nt)
#pragma unroll
    for (int r = 0; r < 8; ++r)
      out[(size_t)(M0 + r + 8 * half) * Dc + (N0 + nt * 16 + l16)] = acc[nt][r];
}

// ---------------------------------------------------------------------------
// Host-side launch.
// Workspace layout (bytes, all sizes 256B-aligned by construction):
//   Xbf  (B*T*D bf16)          16.78 MB
//   Wqb/Wkb/Wvb/Wob (bf16)      8.39/2.10/2.10/8.39 MB
//   Qf/Kf (fp32 head layout)   33.55/8.39 MB
//   Qb/Kb/Vb/Ob (bf16 head)    16.78/4.19/4.19/16.78 MB      (~122 MB total)
// ---------------------------------------------------------------------------
extern "C" void kernel_launch(void* const* d_in, const int* in_sizes, int n_in,
                              void* d_out, int out_size, void* d_ws, size_t ws_size,
                              hipStream_t stream) {
  (void)in_sizes; (void)n_in; (void)out_size; (void)ws_size;
  const float* hs   = (const float*)d_in[0];
  const float* cosT = (const float*)d_in[1];
  const float* sinT = (const float*)d_in[2];
  const float* Wq   = (const float*)d_in[3];
  const float* Wk   = (const float*)d_in[4];
  const float* Wv   = (const float*)d_in[5];
  const float* Wo   = (const float*)d_in[6];
  float* out = (float*)d_out;

  const size_t nX  = (size_t)Bc * Tc * Dc;          // 8.39M
  const size_t nWq = (size_t)NHc * HDc * Dc;        // 4.19M
  const size_t nWk = (size_t)NKVc * HDc * Dc;       // 1.05M
  const size_t nQ  = (size_t)Bc * NHc * Tc * HDc;   // 8.39M
  const size_t nK  = (size_t)Bc * NKVc * Tc * HDc;  // 2.10M

  char* p = (char*)d_ws;
  __bf16* Xbf = (__bf16*)p;            p += nX  * sizeof(__bf16);
  __bf16* Wqb = (__bf16*)p;            p += nWq * sizeof(__bf16);
  __bf16* Wkb = (__bf16*)p;            p += nWk * sizeof(__bf16);
  __bf16* Wvb = (__bf16*)p;            p += nWk * sizeof(__bf16);
  __bf16* Wob = (__bf16*)p;            p += nWq * sizeof(__bf16);
  float*  Qf  = (float*)p;             p += nQ  * sizeof(float);
  float*  Kf  = (float*)p;             p += nK  * sizeof(float);
  __bf16* Qb  = (__bf16*)p;            p += nQ  * sizeof(__bf16);
  __bf16* Kb  = (__bf16*)p;            p += nK  * sizeof(__bf16);
  __bf16* Vb  = (__bf16*)p;            p += nK  * sizeof(__bf16);
  __bf16* Ob  = (__bf16*)p;

  dim3 blk(256);
  const float qscale = 0.08838834764831845f;   // 1/sqrt(128)

  // Pack activations + weights to bf16.
  pack_bf16<<<dim3((unsigned)(nX  / 1024)), blk, 0, stream>>>(hs, Xbf, (int)nX);
  pack_bf16<<<dim3((unsigned)(nWq / 1024)), blk, 0, stream>>>(Wq, Wqb, (int)nWq);
  pack_bf16<<<dim3((unsigned)(nWk / 1024)), blk, 0, stream>>>(Wk, Wkb, (int)nWk);
  pack_bf16<<<dim3((unsigned)(nWk / 1024)), blk, 0, stream>>>(Wv, Wvb, (int)nWk);
  pack_bf16<<<dim3((unsigned)(nWq / 1024)), blk, 0, stream>>>(Wo, Wob, (int)nWq);

  const int Mtiles = (Bc * Tc) / 16;           // 256

  // Projections: Q,K -> fp32 (RoPE/RMS input), V -> bf16 directly.
  gemm_proj<false><<<dim3(Mtiles, (NHc * HDc) / 512), blk, 0, stream>>>(
      Xbf, Wqb, Qf, nullptr, NHc);
  gemm_proj<false><<<dim3(Mtiles, (NKVc * HDc) / 512), blk, 0, stream>>>(
      Xbf, Wkb, Kf, nullptr, NKVc);
  gemm_proj<true><<<dim3(Mtiles, (NKVc * HDc) / 512), blk, 0, stream>>>(
      Xbf, Wvb, nullptr, Vb, NKVc);

  // RoPE + RMSNorm, writing bf16 (Q pre-scaled by 1/sqrt(hd)).
  rope_rms_bf<<<dim3(Bc * NHc * Tc),  dim3(128), 0, stream>>>(Qf, Qb, cosT, sinT, qscale);
  rope_rms_bf<<<dim3(Bc * NKVc * Tc), dim3(128), 0, stream>>>(Kf, Kb, cosT, sinT, 1.0f);

  // Flash attention with TDM-staged KV tiles.
  attention<<<dim3(Bc * NHc * (Tc / 128)), blk, 0, stream>>>(Qb, Kb, Vb, Ob);

  // Output projection.
  gemm_out<<<dim3(Mtiles, Dc / 512), blk, 0, stream>>>(Ob, Wob, out);
}
